// VQ_VAE_88923002896789
// MI455X (gfx1250) — matrix operand order repaired and checked by reading, not compiled, not run on hardware
//
#include <hip/hip_runtime.h>
#include <hip/hip_bf16.h>

// ---------------------------------------------------------------------------
// Residual VQ (4 stages, V=1024, D=64) for gfx1250.
// bf16 WMMA distance matmul, async-to-LDS double-buffered B fragments,
// 32 rows per wave (two A tile pairs) to halve B-bytes per FLOP.
// latent: [32,4096,64] f32 ; emb: [4,1024,64] f32
// out: [1 + 131072*64] f32 (loss, quantised)
// ---------------------------------------------------------------------------

typedef __attribute__((ext_vector_type(16))) __bf16          v16bf;
typedef __attribute__((ext_vector_type(8)))  float           v8f;
typedef __attribute__((ext_vector_type(16))) unsigned short  u16x16;

#define HQ      4
#define VCODES  1024
#define DDIM    64
#define MTOT    (32 * 4096)

#define CHUNKS        64                         // 16-code chunks per stage
#define TILE_CHUNKS   8                          // chunks per LDS tile
#define NTILES        (CHUNKS / TILE_CHUNKS)     // 8 tiles per stage
#define CHUNK_HALVES  (2 * 32 * 16)              // 1024 bf16 = 2 KB
#define TILE_HALVES   (TILE_CHUNKS * CHUNK_HALVES)   // 8192 bf16 = 16 KB
#define WB_STAGE_HALVES (CHUNKS * CHUNK_HALVES)
#define WB_TOT_HALVES   (HQ * WB_STAGE_HALVES)   // 512 KB in d_ws

#if defined(__has_builtin)
#if __has_builtin(__builtin_amdgcn_s_wait_asynccnt)
#define HAVE_WAITASYNC_BUILTIN 1
#endif
#endif

// Async global->LDS 16B copy (CDNA5). VDST = LDS byte offset (addr[31:0]),
// VADDR = 64-bit global address. Tracked by ASYNCcnt.
__device__ __forceinline__ void async_cp16(const void* g, void* l) {
    unsigned lds = (unsigned)(unsigned long long)l;   // LDS offset = addr[31:0]
    asm volatile("global_load_async_to_lds_b128 %0, %1, off"
                 :: "v"(lds), "v"(g) : "memory");
}
__device__ __forceinline__ void wait_async_le4() {
#if defined(HAVE_WAITASYNC_BUILTIN)
    __builtin_amdgcn_s_wait_asynccnt(4);
#else
    asm volatile("s_wait_asynccnt 0x4" ::: "memory");
#endif
}
__device__ __forceinline__ void wait_async_le0() {
#if defined(HAVE_WAITASYNC_BUILTIN)
    __builtin_amdgcn_s_wait_asynccnt(0);
#else
    asm volatile("s_wait_asynccnt 0x0" ::: "memory");
#endif
}

__device__ __forceinline__ unsigned short f2bf(float f) {
    union { float f; unsigned u; } x; x.f = f;
    unsigned u = x.u;
    u += 0x7FFFu + ((u >> 16) & 1u);   // round-to-nearest-even
    return (unsigned short)(u >> 16);
}

// Pre-swizzle codebooks into WMMA B-fragment layout (bf16) + per-code |w|^2.
__global__ void vq_prep_kernel(const float* __restrict__ emb,
                               unsigned short* __restrict__ wb,
                               float* __restrict__ wnorm) {
    int tid = blockIdx.x * blockDim.x + threadIdx.x;
    if (tid < WB_TOT_HALVES) {
        int j = tid & 15;            // half within fragment
        int L = (tid >> 4) & 31;     // lane
        int f = (tid >> 9) & 1;      // K fragment (0: K=0..31, 1: K=32..63)
        int c = (tid >> 10) & 63;    // 16-code chunk
        int s = tid >> 16;           // stage
        int v = c * 16 + (L & 15);   // code (B column N = lane%16)
        int k = f * 32 + ((j < 8) ? j : j + 8) + ((L < 16) ? 0 : 8);  // ISA 16-bit B layout
        wb[tid] = f2bf(emb[((size_t)s * VCODES + v) * DDIM + k]);
    }
    if (tid < HQ * VCODES) {
        const float* w = emb + (size_t)tid * DDIM;
        float acc = 0.f;
        #pragma unroll
        for (int k = 0; k < DDIM; ++k) { float x = w[k]; acc += x * x; }
        wnorm[tid] = acc;
    }
}

// Issue one 16KB tile (4 x b128 per thread) of B fragments into LDS.
__device__ __forceinline__ void stage_tile(const unsigned short* gtile,
                                           unsigned short* ltile, int tid) {
    const char* g = (const char*)gtile + tid * 16;
    char* l = (char*)ltile + tid * 16;
    #pragma unroll
    for (int rnd = 0; rnd < 4; ++rnd)
        async_cp16(g + rnd * 4096, l + rnd * 4096);
}

__global__ __launch_bounds__(256) void vq_main_kernel(
        const float* __restrict__ latent,
        const float* __restrict__ emb,
        const unsigned short* __restrict__ wb,
        const float* __restrict__ wnorm,
        float* __restrict__ out) {
    __shared__ unsigned short sB[2][TILE_HALVES];   // 32 KB double buffer

    const int tid  = threadIdx.x;
    const int lane = tid & 31;
    const int wave = tid >> 5;
    const int rowBase = blockIdx.x * 256 + wave * 32;   // 32 rows per wave
    const int m   = lane & 15;
    const int off = (lane >= 16) ? 8 : 0;               // A-layout K offset
    const int gm0 = rowBase + m;                        // tile-0 row
    const int gm1 = rowBase + 16 + m;                   // tile-1 row

    // Residuals in A-fragment element order: r[f][j] = row[f*32+((j<8)?j:j+8)+off]
    float r0[2][16], r1[2][16];
    const float* lrow0 = latent + (size_t)gm0 * DDIM;
    const float* lrow1 = latent + (size_t)gm1 * DDIM;
    #pragma unroll
    for (int f = 0; f < 2; ++f)
        #pragma unroll
        for (int j = 0; j < 16; ++j) {
            int k = f * 32 + ((j < 8) ? j : j + 8) + off;
            r0[f][j] = lrow0[k];
            r1[f][j] = lrow1[k];
        }

    float lossAcc = 0.f;

    for (int s = 0; s < HQ; ++s) {
        // bf16 A fragments for both 16-row tiles
        u16x16 u0, u1, u2, u3;
        #pragma unroll
        for (int j = 0; j < 16; ++j) {
            u0[j] = f2bf(r0[0][j]); u1[j] = f2bf(r0[1][j]);
            u2[j] = f2bf(r1[0][j]); u3[j] = f2bf(r1[1][j]);
        }
        v16bf a0 = __builtin_bit_cast(v16bf, u0);
        v16bf a1 = __builtin_bit_cast(v16bf, u1);
        v16bf a2 = __builtin_bit_cast(v16bf, u2);
        v16bf a3 = __builtin_bit_cast(v16bf, u3);

        float best0[8], best1[8]; int bidx0[8], bidx1[8];
        #pragma unroll
        for (int e = 0; e < 8; ++e) {
            best0[e] = 3.4e38f; bidx0[e] = 0;
            best1[e] = 3.4e38f; bidx1[e] = 0;
        }

        const unsigned short* wbs = wb + (size_t)s * WB_STAGE_HALVES;
        const float* wns = wnorm + s * VCODES;

        stage_tile(wbs, sB[0], tid);   // prefetch tile 0

        for (int t = 0; t < NTILES; ++t) {
            if (t + 1 < NTILES) {
                stage_tile(wbs + (size_t)(t + 1) * TILE_HALVES, sB[(t + 1) & 1], tid);
                wait_async_le4();      // tile t's 4 loads complete (in-order)
            } else {
                wait_async_le0();
            }
            __syncthreads();           // tile t visible to all waves

            const unsigned short* buf = sB[t & 1];
            #pragma unroll 2
            for (int cc = 0; cc < TILE_CHUNKS; ++cc) {
                v16bf b0 = *(const v16bf*)(buf + ((cc * 2 + 0) * 32 + lane) * 16);
                v16bf b1 = *(const v16bf*)(buf + ((cc * 2 + 1) * 32 + lane) * 16);
                v8f acc0 = {}, acc1 = {};
                acc0 = __builtin_amdgcn_wmma_f32_16x16x32_bf16(false, a0, false, b0,
                                                               (short)0, acc0, false, false);
                acc0 = __builtin_amdgcn_wmma_f32_16x16x32_bf16(false, a1, false, b1,
                                                               (short)0, acc0, false, false);
                acc1 = __builtin_amdgcn_wmma_f32_16x16x32_bf16(false, a2, false, b0,
                                                               (short)0, acc1, false, false);
                acc1 = __builtin_amdgcn_wmma_f32_16x16x32_bf16(false, a3, false, b1,
                                                               (short)0, acc1, false, false);
                int v = (t * TILE_CHUNKS + cc) * 16 + m;
                float wn = wns[v];
                #pragma unroll
                for (int e = 0; e < 8; ++e) {
                    float sc0 = wn - 2.f * acc0[e];
                    float sc1 = wn - 2.f * acc1[e];
                    bool g0 = sc0 < best0[e];
                    bool g1 = sc1 < best1[e];
                    best0[e] = g0 ? sc0 : best0[e];  bidx0[e] = g0 ? v : bidx0[e];
                    best1[e] = g1 ? sc1 : best1[e];  bidx1[e] = g1 ? v : bidx1[e];
                }
            }
            __syncthreads();           // done reading buf before it is re-filled
        }

        // Finish argmin across the 16 lanes of each half (tie -> lower index)
        #pragma unroll
        for (int e = 0; e < 8; ++e) {
            #pragma unroll
            for (int d = 1; d < 16; d <<= 1) {
                float os0 = __shfl_xor(best0[e], d, 32);
                int   oi0 = __shfl_xor(bidx0[e], d, 32);
                if (os0 < best0[e] || (os0 == best0[e] && oi0 < bidx0[e])) {
                    best0[e] = os0; bidx0[e] = oi0;
                }
                float os1 = __shfl_xor(best1[e], d, 32);
                int   oi1 = __shfl_xor(bidx1[e], d, 32);
                if (os1 < best1[e] || (os1 == best1[e] && oi1 < bidx1[e])) {
                    best1[e] = os1; bidx1[e] = oi1;
                }
            }
        }

        // Each lane needs the winning code of its own row m (uniform broadcasts)
        int myIdx0 = 0, myIdx1 = 0;
        #pragma unroll
        for (int rr = 0; rr < 16; ++rr) {
            int iv0 = __shfl(bidx0[rr & 7], (rr < 8) ? 0 : 16, 32);
            int iv1 = __shfl(bidx1[rr & 7], (rr < 8) ? 0 : 16, 32);
            if (rr == m) { myIdx0 = iv0; myIdx1 = iv1; }
        }

        // Gather winning codes in f32 (L2-hot), update loss and residuals
        const float* q0 = emb + ((size_t)s * VCODES + myIdx0) * DDIM;
        const float* q1 = emb + ((size_t)s * VCODES + myIdx1) * DDIM;
        #pragma unroll
        for (int f = 0; f < 2; ++f)
            #pragma unroll
            for (int j = 0; j < 16; ++j) {
                int k = f * 32 + ((j < 8) ? j : j + 8) + off;
                float qa = q0[k], qb = q1[k];
                float da = qa - r0[f][j], db = qb - r1[f][j];
                lossAcc += da * da + db * db;
                r0[f][j] -= qa;
                r1[f][j] -= qb;
            }
    }

    // quantised == code_sum == latent - r_final (f32); latent is L2-resident
    float* qo0 = out + 1 + (size_t)gm0 * DDIM;
    float* qo1 = out + 1 + (size_t)gm1 * DDIM;
    #pragma unroll
    for (int f = 0; f < 2; ++f)
        #pragma unroll
        for (int j = 0; j < 16; ++j) {
            int k = f * 32 + ((j < 8) ? j : j + 8) + off;
            qo0[k] = lrow0[k] - r0[f][j];
            qo1[k] = lrow1[k] - r1[f][j];
        }

    // loss = sum_stages 1.25 * mean((q - r)^2); divisor = 32*4096*64
    #pragma unroll
    for (int d = 1; d < 32; d <<= 1) lossAcc += __shfl_xor(lossAcc, d, 32);
    if (lane == 0) atomicAdd(out, lossAcc * (1.25f / 8388608.f));
}

extern "C" void kernel_launch(void* const* d_in, const int* in_sizes, int n_in,
                              void* d_out, int out_size, void* d_ws, size_t ws_size,
                              hipStream_t stream) {
    const float* latent = (const float*)d_in[0];   // [32,4096,64] f32
    const float* emb    = (const float*)d_in[1];   // [4,1024,64]  f32
    float* out = (float*)d_out;                    // [1 + 8388608] f32

    unsigned short* wb = (unsigned short*)d_ws;                     // 512 KB
    float* wnorm = (float*)((char*)d_ws + WB_TOT_HALVES * sizeof(unsigned short)); // 16 KB

    (void)hipMemsetAsync(out, 0, sizeof(float), stream);  // loss accumulator

    vq_prep_kernel<<<(WB_TOT_HALVES + 255) / 256, 256, 0, stream>>>(emb, wb, wnorm);

    // 131072 rows / (8 waves * 32 rows) = 512 blocks of 256 threads (8 wave32)
    vq_main_kernel<<<MTOT / 256, 256, 0, stream>>>(latent, emb, wb, wnorm, out);
}
